// MultiHeadedAttention_8770323218442
// MI455X (gfx1250) — matrix operand order repaired and checked by reading, not compiled, not run on hardware
//
#include <hip/hip_runtime.h>
#include <hip/hip_bf16.h>
#include <stdint.h>

// Problem constants (match reference)
#define EMBED 1024
#define HEADS 16
#define HDIM  64
#define BATCH 2
#define SEQ   2048
#define MTOT  (BATCH * SEQ)   // 4096 rows

typedef __attribute__((ext_vector_type(16))) __bf16 v16bf;
typedef __attribute__((ext_vector_type(8)))  float  v8f;
#if __has_builtin(__builtin_amdgcn_cvt_pk_bf16_f32)
typedef __attribute__((ext_vector_type(2)))  __bf16 v2bf;
#endif

#define AS3L __attribute__((address_space(3)))
#define HAVE_ASYNC_LDS 1
#if __has_builtin(__builtin_amdgcn_tensor_load_to_lds) && \
    __has_builtin(__builtin_amdgcn_s_wait_tensorcnt)
#define HAVE_TDM 1
typedef unsigned int u32x4 __attribute__((ext_vector_type(4)));
typedef int          i32x8 __attribute__((ext_vector_type(8)));
typedef int          i32x4 __attribute__((ext_vector_type(4)));
#endif

union FragA {            // 16x32 bf16 A/B operand fragment (8 VGPRs)
  v16bf    v;
  uint16_t u[16];
  uint4    q[2];
};
union FragC {            // 16x16 f32 C/D fragment (8 VGPRs)
  v8f   v;
  float f[8];
};

__device__ __forceinline__ uint16_t f2bf_sw(float x) {
  uint32_t u = __builtin_bit_cast(uint32_t, x);
  u += 0x7FFFu + ((u >> 16) & 1u);
  return (uint16_t)(u >> 16);
}
__device__ __forceinline__ uint32_t pack_bf16(float lo, float hi) {
#if __has_builtin(__builtin_amdgcn_cvt_pk_bf16_f32)
  v2bf p = __builtin_amdgcn_cvt_pk_bf16_f32(lo, hi);
  return __builtin_bit_cast(uint32_t, p);
#else
  return (uint32_t)f2bf_sw(lo) | ((uint32_t)f2bf_sw(hi) << 16);
#endif
}
__device__ __forceinline__ uint16_t f2bf(float x) {
#if __has_builtin(__builtin_amdgcn_cvt_pk_bf16_f32)
  return (uint16_t)(pack_bf16(x, x) & 0xFFFFu);
#else
  return f2bf_sw(x);
#endif
}

__device__ __forceinline__ v8f wmma_bf16(v16bf a, v16bf b, v8f c) {
  return __builtin_amdgcn_wmma_f32_16x16x32_bf16(
      false, a, false, b, (short)0, c, false, false);
}

__device__ __forceinline__ uint32_t lds_addr_of(const void* p) {
  return (uint32_t)(uintptr_t)((AS3L void*)p);
}

// ---- per-lane async global->LDS copy (ASYNCcnt), ISA §15.18 ----
__device__ __forceinline__ void async_b128(const void* g, void* l) {
  const uint32_t loff  = lds_addr_of(l);
  const uint64_t gaddr = (uint64_t)(uintptr_t)g;
  asm volatile("global_load_async_to_lds_b128 %0, %1, off"
               :: "v"(loff), "v"(gaddr)
               : "memory");
}
__device__ __forceinline__ void wait_async0() {
#if __has_builtin(__builtin_amdgcn_s_wait_asynccnt)
  __builtin_amdgcn_s_wait_asynccnt(0);
#else
  asm volatile("s_wait_asynccnt 0x0" ::: "memory");
#endif
}

#ifdef HAVE_TDM
// ---- Tensor Data Mover: 2D tile (tile_rows x 32 bf16) -> LDS, row pad 16B ----
// D# per ISA §8.3/8.4: data_size=2B, tensor_dim0=1024, dim0_stride=1024,
// tile_dim0=32 elems (64B row), pad_interval=3 (16 DW) + pad_amount=3 (4 DW)
// -> LDS row stride 80B == LDS_STRIDE*2.
__device__ __forceinline__ void tdm_load_tile(
    const void* gtile, uint32_t lds_byte, uint32_t tensor_rows,
    uint32_t tile_rows) {
  const uint64_t ga = (uint64_t)(uintptr_t)gtile;
  u32x4 g0;
  g0.x = 1u;                                   // count=1 (valid user D#)
  g0.y = lds_byte;                             // lds_addr
  g0.z = (uint32_t)ga;                         // global_addr[31:0]
  g0.w = (uint32_t)((ga >> 32) & 0x01FFFFFFu)  // global_addr[56:32]
         | (2u << 30);                         // type=2 ("image")
  i32x8 g1;
  g1[0] = (1 << 16)                            // data_size = 2 bytes
        | (1 << 20)                            // pad_enable
        | (3 << 22)                            // pad_interval: 16 DWORDs
        | (3 << 25);                           // pad_amount: 4 DWORDs (16B)
  g1[1] = (int)(1024u << 16);                  // tensor_dim0 = 1024 (lo16)
  g1[2] = (int)(tensor_rows << 16);            // dim0 hi=0 | tensor_dim1 lo16
  g1[3] = (int)(32u << 16);                    // dim1 hi=0 | tile_dim0 = 32
  g1[4] = (int)tile_rows;                      // tile_dim1 | tile_dim2=0
  g1[5] = 1024;                                // tensor_dim0_stride lo32
  g1[6] = 0;                                   // stride hi | dim1_stride lo
  g1[7] = 0;
  const i32x4 z4 = {0, 0, 0, 0};               // groups 2/3 unused (2D tensor)
  const i32x8 z8 = {0, 0, 0, 0, 0, 0, 0, 0};   // extra group (6-arg toolchain)
  __builtin_amdgcn_tensor_load_to_lds(g0, g1, z4, z4, z8, 0);
}
__device__ __forceinline__ void wait_tensor0() {
  __builtin_amdgcn_s_wait_tensorcnt(0);
}
#endif

// ---------------------------------------------------------------------------
// Bulk fp32 -> bf16 conversion (one pass; keeps conversions out of GEMM loops)
// ---------------------------------------------------------------------------
__global__ __launch_bounds__(256) void cvt_f32_bf16(
    const float* __restrict__ src, uint16_t* __restrict__ dst, int n) {
  const int i = (blockIdx.x * 256 + threadIdx.x) * 8;
  if (i + 8 > n) return;
  const float4 a = *(const float4*)(src + i);
  const float4 b = *(const float4*)(src + i + 4);
  uint4 o;
  o.x = pack_bf16(a.x, a.y);
  o.y = pack_bf16(a.z, a.w);
  o.z = pack_bf16(b.x, b.y);
  o.w = pack_bf16(b.z, b.w);
  *(uint4*)(dst + i) = o;
}

// ---------------------------------------------------------------------------
// Unified LDS-tiled GEMM:  Y = A(bf16)[M,1024] @ W(bf16)[1024,1024]^T + bias
// Block tile 128(M) x 64(N), BK=32, 8 waves of 32x32, ping-pong LDS buffers.
// Staging: TDM TENSOR_LOAD_TO_LDS (TENSORcnt, warp 0 issues, pad_enable gives
// the padded LDS stride) when available; else per-lane async loads (ASYNCcnt).
// out_mode: 0 -> bf16 [B,H,S,D] (Q,K); 1 -> bf16 [B,H,D,S] (V^T); 2 -> f32.
// ---------------------------------------------------------------------------
#define BM 128
#define BN 64
#define BK 32
#define LDS_STRIDE 40   // BK + 8 pad: 80-byte rows -> conflict-free b128 reads

__global__ __launch_bounds__(256) void gemm_bf16(
    const uint16_t* __restrict__ A, const uint16_t* __restrict__ Wb,
    const float* __restrict__ bias, void* __restrict__ Yv, int out_mode) {
  __shared__ __align__(16) uint16_t sA[2][BM * LDS_STRIDE];  // 2 x 10 KB
  __shared__ __align__(16) uint16_t sB[2][BN * LDS_STRIDE];  // 2 x  5 KB

  const int tid  = threadIdx.x;
  const int lane = tid & 31;
  const int warp = tid >> 5;
  const int nBlkM = MTOT / BM;                    // 32
  const int m0 = (blockIdx.x % nBlkM) * BM;
  const int n0 = (blockIdx.x / nBlkM) * BN;
  const int wm0 = (warp & 3) * 32;                // wave tile in block: 4x2
  const int wn0 = (warp >> 2) * 32;
  const int r  = lane & 15;
  const int hh = lane >> 4;

  FragC acc[2][2];
#pragma unroll
  for (int mi = 0; mi < 2; ++mi)
#pragma unroll
    for (int ni = 0; ni < 2; ++ni)
#pragma unroll
      for (int e = 0; e < 8; ++e) acc[mi][ni].f[e] = 0.0f;

#ifdef HAVE_TDM
  // ---------------- TDM double-buffered pipeline ----------------
  const uint16_t* gA = A  + (size_t)m0 * EMBED;
  const uint16_t* gB = Wb + (size_t)n0 * EMBED;
  const uint32_t ldsA0 = lds_addr_of(&sA[0][0]), ldsA1 = lds_addr_of(&sA[1][0]);
  const uint32_t ldsB0 = lds_addr_of(&sB[0][0]), ldsB1 = lds_addr_of(&sB[1][0]);

  if (warp == 0) {
    tdm_load_tile(gA, ldsA0, MTOT, BM);
    tdm_load_tile(gB, ldsB0, EMBED, BN);
    wait_tensor0();
  }
  __syncthreads();

  int cur = 0;
  for (int k0 = 0; k0 < EMBED; k0 += BK) {
    const bool more = (k0 + BK) < EMBED;
    if (more && warp == 0) {
      const int kn = k0 + BK;
      tdm_load_tile(gA + kn, cur ? ldsA0 : ldsA1, MTOT, BM);
      tdm_load_tile(gB + kn, cur ? ldsB0 : ldsB1, EMBED, BN);
    }
#else
  // ---------------- per-lane async double-buffered pipeline ----------------
  const int srow = tid >> 2;                      // 0..63
  const int scol = (tid & 3) * 8;                 // 0,8,16,24
  const uint16_t* gA0 = A  + (size_t)(m0 + srow) * EMBED + scol;
  const uint16_t* gA1 = A  + (size_t)(m0 + srow + 64) * EMBED + scol;
  const uint16_t* gB0 = Wb + (size_t)(n0 + srow) * EMBED + scol;
  const int lofs = srow * LDS_STRIDE + scol;

  async_b128(gA0, &sA[0][lofs]);
  async_b128(gA1, &sA[0][lofs + 64 * LDS_STRIDE]);
  async_b128(gB0, &sB[0][lofs]);
  wait_async0();
  __syncthreads();

  int cur = 0;
  for (int k0 = 0; k0 < EMBED; k0 += BK) {
    const bool more = (k0 + BK) < EMBED;
    if (more) {
      const int kn = k0 + BK;
      async_b128(gA0 + kn, &sA[cur ^ 1][lofs]);
      async_b128(gA1 + kn, &sA[cur ^ 1][lofs + 64 * LDS_STRIDE]);
      async_b128(gB0 + kn, &sB[cur ^ 1][lofs]);
    }
#endif

    FragA af[2], bf[2];
#pragma unroll
    for (int mi = 0; mi < 2; ++mi) {
      const uint16_t* p = &sA[cur][(wm0 + 16 * mi + r) * LDS_STRIDE];
      af[mi].q[0] = *(const uint4*)(p + 8 * hh);
      af[mi].q[1] = *(const uint4*)(p + 16 + 8 * hh);
    }
#pragma unroll
    for (int ni = 0; ni < 2; ++ni) {
      const uint16_t* p = &sB[cur][(wn0 + 16 * ni + r) * LDS_STRIDE];
      bf[ni].q[0] = *(const uint4*)(p + 8 * hh);
      bf[ni].q[1] = *(const uint4*)(p + 16 + 8 * hh);
    }
#pragma unroll
    for (int mi = 0; mi < 2; ++mi)
#pragma unroll
      for (int ni = 0; ni < 2; ++ni)
        acc[mi][ni].v = wmma_bf16(af[mi].v, bf[ni].v, acc[mi][ni].v);

#ifdef HAVE_TDM
    if (more && warp == 0) wait_tensor0();
#else
    if (more) wait_async0();
#endif
    __syncthreads();
    cur ^= 1;
  }

  // Epilogue
#pragma unroll
  for (int mi = 0; mi < 2; ++mi)
#pragma unroll
    for (int ni = 0; ni < 2; ++ni) {
      const int col  = n0 + wn0 + 16 * ni + r;    // C frag: col = lane&15
      const float bv = bias[col];
      const int hcol = col >> 6;
      const int dcol = col & (HDIM - 1);
#pragma unroll
      for (int j = 0; j < 8; ++j) {
        const int row = m0 + wm0 + 16 * mi + 8 * hh + j;  // row = 8h + j
        const float vo = acc[mi][ni].f[j] + bv;
        if (out_mode == 2) {
          ((float*)Yv)[(size_t)row * EMBED + col] = vo;
        } else {
          const int bb = row >> 11;
          const int s  = row & (SEQ - 1);
          const size_t idx = (out_mode == 1)
              ? ((size_t)(bb * HEADS + hcol) * HDIM + dcol) * SEQ + s
              : ((size_t)(bb * HEADS + hcol) * SEQ + s) * HDIM + dcol;
          ((uint16_t*)Yv)[idx] = f2bf(vo);
        }
      }
    }
}

// ---------------------------------------------------------------------------
// Flash-style causal attention. One wave owns 16 query rows of one (b,h).
// ---------------------------------------------------------------------------
__global__ __launch_bounds__(256) void flash_attn(
    const uint16_t* __restrict__ Qb, const uint16_t* __restrict__ Kb,
    const uint16_t* __restrict__ Vt, uint16_t* __restrict__ Ob) {
  __shared__ __align__(16) uint16_t plds[8][16 * 32];

  const int lane = threadIdx.x & 31;
  const int warp = threadIdx.x >> 5;
  const int wid  = blockIdx.x * 8 + warp;
  const int qt   = wid % (SEQ / 16);
  const int h    = (wid / (SEQ / 16)) % HEADS;
  const int bb   = wid / ((SEQ / 16) * HEADS);
  const int qbase = qt * 16;
  const int r  = lane & 15;
  const int hh = lane >> 4;

  const uint16_t* Qp = Qb + (size_t)(bb * HEADS + h) * SEQ * HDIM;
  const uint16_t* Kp = Kb + (size_t)(bb * HEADS + h) * SEQ * HDIM;
  const uint16_t* Vp = Vt + (size_t)(bb * HEADS + h) * HDIM * SEQ;
  uint16_t* lp = plds[warp];

  FragA qf[2];
#pragma unroll
  for (int kd = 0; kd < 2; ++kd) {
    const uint16_t* qp = Qp + (size_t)(qbase + r) * HDIM + 32 * kd;
    qf[kd].q[0] = *(const uint4*)(qp + 8 * hh);
    qf[kd].q[1] = *(const uint4*)(qp + 16 + 8 * hh);
  }

  FragC o[4];
#pragma unroll
  for (int dt = 0; dt < 4; ++dt)
#pragma unroll
    for (int e = 0; e < 8; ++e) o[dt].f[e] = 0.0f;
  float mrow[8], lrow[8];
#pragma unroll
  for (int j = 0; j < 8; ++j) { mrow[j] = -3.0e38f; lrow[j] = 0.0f; }

  const int nkb = (qbase + 16 + 31) >> 5;   // causal KV block bound
  for (int kb = 0; kb < nkb; ++kb) {
    const int kv0 = kb * 32;

    FragC sfr[2];
#pragma unroll
    for (int nt = 0; nt < 2; ++nt)
#pragma unroll
      for (int e = 0; e < 8; ++e) sfr[nt].f[e] = 0.0f;
#pragma unroll
    for (int nt = 0; nt < 2; ++nt)
#pragma unroll
      for (int kd = 0; kd < 2; ++kd) {
        FragA bk;
        const uint16_t* kp =
            Kp + (size_t)(kv0 + nt * 16 + r) * HDIM + 32 * kd;
        bk.q[0] = *(const uint4*)(kp + 8 * hh);
        bk.q[1] = *(const uint4*)(kp + 16 + 8 * hh);
        sfr[nt].v = wmma_bf16(qf[kd].v, bk.v, sfr[nt].v);
      }

    float sv[2][8];
#pragma unroll
    for (int j = 0; j < 8; ++j) {
      const int row = qbase + 8 * hh + j;
#pragma unroll
      for (int nt = 0; nt < 2; ++nt) {
        const int col = kv0 + nt * 16 + r;
        float e = sfr[nt].f[j] * 0.125f;          // 1/sqrt(64)
        if (col > row) e = -3.0e38f;              // causal mask
        sv[nt][j] = e;
      }
      float lm = fmaxf(sv[0][j], sv[1][j]);
#pragma unroll
      for (int off = 1; off < 16; off <<= 1)
        lm = fmaxf(lm, __shfl_xor(lm, off, 32));
      const float mnew  = fmaxf(mrow[j], lm);
      const float alpha = __expf(mrow[j] - mnew);
      float psum = 0.0f;
#pragma unroll
      for (int nt = 0; nt < 2; ++nt) {
        const float p = __expf(sv[nt][j] - mnew);
        sv[nt][j] = p;
        psum += p;
      }
#pragma unroll
      for (int off = 1; off < 16; off <<= 1) psum += __shfl_xor(psum, off, 32);
      lrow[j] = lrow[j] * alpha + psum;
      mrow[j] = mnew;
#pragma unroll
      for (int dt = 0; dt < 4; ++dt) o[dt].f[j] *= alpha;
    }

    // P: C-layout -> LDS -> A-layout (bf16)
#pragma unroll
    for (int nt = 0; nt < 2; ++nt)
#pragma unroll
      for (int j = 0; j < 8; ++j)
        lp[(8 * hh + j) * 32 + nt * 16 + r] = f2bf(sv[nt][j]);
    FragA pf;
    pf.q[0] = *(const uint4*)(lp + r * 32 + 8 * hh);
    pf.q[1] = *(const uint4*)(lp + r * 32 + 16 + 8 * hh);

#pragma unroll
    for (int dt = 0; dt < 4; ++dt) {
      FragA bv;
      const uint16_t* vp = Vp + (size_t)(dt * 16 + r) * SEQ + kv0;
      bv.q[0] = *(const uint4*)(vp + 8 * hh);
      bv.q[1] = *(const uint4*)(vp + 16 + 8 * hh);
      o[dt].v = wmma_bf16(pf.v, bv.v, o[dt].v);
    }
  }

#pragma unroll
  for (int j = 0; j < 8; ++j) {
    const float inv = 1.0f / lrow[j];
    const int row   = qbase + 8 * hh + j;
    uint16_t* op = Ob + ((size_t)bb * SEQ + row) * EMBED + h * HDIM;
#pragma unroll
    for (int dt = 0; dt < 4; ++dt) op[dt * 16 + r] = f2bf(o[dt].f[j] * inv);
  }
}

// ---------------------------------------------------------------------------
extern "C" void kernel_launch(void* const* d_in, const int* in_sizes, int n_in,
                              void* d_out, int out_size, void* d_ws,
                              size_t ws_size, hipStream_t stream) {
  (void)in_sizes; (void)n_in; (void)out_size; (void)ws_size;
  const float* query  = (const float*)d_in[0];
  const float* key    = (const float*)d_in[1];
  const float* values = (const float*)d_in[2];
  // d_in[3]: causal mask (tril) — hardcoded in flash_attn
  const float* Wq = (const float*)d_in[4];
  const float* bq = (const float*)d_in[5];
  const float* Wk = (const float*)d_in[6];
  const float* bk = (const float*)d_in[7];
  const float* Wv = (const float*)d_in[8];
  const float* bv = (const float*)d_in[9];
  const float* Wo = (const float*)d_in[10];
  const float* bo = (const float*)d_in[11];
  float* out = (float*)d_out;

  const size_t actE = (size_t)MTOT * EMBED;   // 4,194,304
  const size_t wE   = (size_t)EMBED * EMBED;  // 1,048,576
  uint16_t* qb = (uint16_t*)d_ws;             // bf16 Q  [B,H,S,D]
  uint16_t* kb = qb + actE;                   // bf16 K  [B,H,S,D]
  uint16_t* vt = kb + actE;                   // bf16 V^T [B,H,D,S]
  uint16_t* ct = vt + actE;                   // bf16 context [B,S,E]
  uint16_t* xq = ct + actE;                   // bf16 activations
  uint16_t* xk = xq + actE;
  uint16_t* xv = xk + actE;
  uint16_t* wq = xv + actE;                   // bf16 weights
  uint16_t* wk = wq + wE;
  uint16_t* wv = wk + wE;
  uint16_t* wo = wv + wE;

  const dim3 blk(256);
  const int cvtA = (int)(actE / (256 * 8));   // 2048 blocks
  const int cvtW = (int)(wE / (256 * 8));     // 512 blocks

  cvt_f32_bf16<<<cvtA, blk, 0, stream>>>(query,  xq, (int)actE);
  cvt_f32_bf16<<<cvtA, blk, 0, stream>>>(key,    xk, (int)actE);
  cvt_f32_bf16<<<cvtA, blk, 0, stream>>>(values, xv, (int)actE);
  cvt_f32_bf16<<<cvtW, blk, 0, stream>>>(Wq, wq, (int)wE);
  cvt_f32_bf16<<<cvtW, blk, 0, stream>>>(Wk, wk, (int)wE);
  cvt_f32_bf16<<<cvtW, blk, 0, stream>>>(Wv, wv, (int)wE);
  cvt_f32_bf16<<<cvtW, blk, 0, stream>>>(Wo, wo, (int)wE);

  const int gemm_blocks = (MTOT / BM) * (EMBED / BN);      // 512
  gemm_bf16<<<gemm_blocks, blk, 0, stream>>>(xq, wq, bq, qb, 0);
  gemm_bf16<<<gemm_blocks, blk, 0, stream>>>(xk, wk, bk, kb, 0);
  gemm_bf16<<<gemm_blocks, blk, 0, stream>>>(xv, wv, bv, vt, 1);

  const int attn_blocks = BATCH * HEADS * (SEQ / 16) / 8;  // 512
  flash_attn<<<attn_blocks, blk, 0, stream>>>(qb, kb, vt, ct);

  gemm_bf16<<<gemm_blocks, blk, 0, stream>>>(ct, wo, bo, out, 2);
}
